// RNTN_51805895524400
// MI455X (gfx1250) — matrix operand order repaired and checked by reading, not compiled, not run on hardware
//
#include <hip/hip_runtime.h>
#include <hip/hip_bf16.h>
#include <math.h>
#include <stdint.h>

// ---------------------------------------------------------------------------
// RNTN forward for MI455X (gfx1250, wave32, WMMA + async-to-LDS).
//
//   xVx[m,k] = sum_{j,l} C[m,j] * V[k,j,l] * C[m,l]
//            = sum_{jl}  G[m,jl] * Vflat[jl,k],   G[m,j*64+l] = C[m,j]*C[m,l]
//
// One K=4096 GEMM per 16-node tile via v_wmma_f32_16x16x32_f16; the c@W
// linear term is folded in as 2 extra K-chunks. V/W^T are pre-converted to
// f16 in global once, then staged to LDS with global_load_async_to_lds_b128.
// ---------------------------------------------------------------------------

typedef __attribute__((ext_vector_type(16))) _Float16 v16h;
typedef __attribute__((ext_vector_type(8)))  _Float16 v8h;
typedef __attribute__((ext_vector_type(4)))  _Float16 v4h;
typedef __attribute__((ext_vector_type(8)))  float    v8f;
typedef __attribute__((ext_vector_type(4)))  float    v4f;

#define LEAF_ROWS   32768            // B*L = 512*64
#define NODES_TOTAL 65024            // 512*127
#define OUT_CLS     5

// LDS layout (dynamic shared), bytes:
#define SMEM_V_BYTES    (32 * 64 * 64 * 2)            // V as f16 [k][j][l]  = 256 KB
#define SMEM_WT_OFF     (SMEM_V_BYTES)
#define SMEM_WT_BYTES   (32 * 64 * 2)                 // W^T as f16 [k][l]   = 4 KB
#define SMEM_C_OFF      (SMEM_WT_OFF + SMEM_WT_BYTES) // per-wave C tiles
#define SMEM_C_PER_WAVE (16 * 64 * 2)                 // 16 nodes x 64 f16   = 2 KB
#define SMEM_TOTAL      (SMEM_C_OFF + 8 * SMEM_C_PER_WAVE)  // 282624 B < 320 KB

// d_ws byte layout: [nodes f32 8,323,072 B][Vh f16 262,144 B][Wt f16 4,096 B]
#define WS_NODES_BYTES  ((size_t)NODES_TOTAL * 32 * 4)
#define WS_VH_OFF       WS_NODES_BYTES
#define WS_WT_OFF       (WS_VH_OFF + (size_t)SMEM_V_BYTES)

// ---------------------------------------------------------------------------
// CDNA5 async copy: global -> LDS, 16 B per lane, tracked by ASYNCcnt.
// lds_off = LDS byte address (flat-pointer low 32 bits), gptr = global addr.
// ---------------------------------------------------------------------------
__device__ __forceinline__ void async_copy_b128(unsigned lds_off, const void* gptr) {
    asm volatile("global_load_async_to_lds_b128 %0, %1, off"
                 :: "v"(lds_off), "v"(gptr)
                 : "memory");
}
__device__ __forceinline__ void wait_async_zero() {
    asm volatile("s_wait_asynccnt 0" ::: "memory");
}

// ---------------------------------------------------------------------------
// One-time prep: V (32*64*64) and W^T (32x64) to f16 in global workspace.
// ---------------------------------------------------------------------------
__global__ void rntn_prep_kernel(const float* __restrict__ V,
                                 const float* __restrict__ W,
                                 _Float16* __restrict__ Vh,
                                 _Float16* __restrict__ Wt) {
    int i = blockIdx.x * blockDim.x + threadIdx.x;   // 131072 threads
    Vh[i] = (_Float16)V[i];
    if (i < 64 * 32) {                               // Wt[k*64+l] = W[l*32+k]
        int l = i >> 5, k = i & 31;
        Wt[k * 64 + l] = (_Float16)W[i];
    }
}

// ---------------------------------------------------------------------------
// Embedding gather: nodes[row, :] = embed[word_ids[row], :]   (float4 lanes)
// ---------------------------------------------------------------------------
__global__ void rntn_gather_kernel(const int* __restrict__ ids,
                                   const float* __restrict__ embed,
                                   float* __restrict__ nodes) {
    int i   = blockIdx.x * blockDim.x + threadIdx.x;   // 32768*8 float4 slots
    int row = i >> 3;
    int c4  = i & 7;
    int id  = ids[row];
    v4f v = *(const v4f*)(embed + (size_t)id * 32 + c4 * 4);
    *(v4f*)(nodes + (size_t)row * 32 + c4 * 4) = v;
}

// ---------------------------------------------------------------------------
// One tree level. Cin = (Nout, 64) contiguous view of parent level,
// Hout = (Nout, 32). Block = 256 threads = 8 wave32s, one 16-node tile/wave.
// ---------------------------------------------------------------------------
__global__ void rntn_level_kernel(const float* __restrict__ Cin,
                                  float* __restrict__ Hout,
                                  const _Float16* __restrict__ Vh_g,
                                  const _Float16* __restrict__ Wt_g,
                                  const float* __restrict__ bias) {
    extern __shared__ char smem[];
    const _Float16* Vh = (const _Float16*)smem;                 // [k][j][l]
    const _Float16* Wt = (const _Float16*)(smem + SMEM_WT_OFF); // [k][l]

    const int tid = threadIdx.x;
    const unsigned lds_base = (unsigned)(uintptr_t)smem;        // LDS aperture: addr[31:0]

    // ---- Async stage V f16 (256 KB) + W^T f16 (4 KB) into LDS ----
    {
        const char* vg = (const char*)Vh_g;
        #pragma unroll 4
        for (int it = 0; it < 64; ++it) {                       // 64*256*16B = 256 KB
            const int off = (it * 256 + tid) * 16;
            async_copy_b128(lds_base + off, vg + off);
        }
        async_copy_b128(lds_base + SMEM_WT_OFF + tid * 16,
                        (const char*)Wt_g + tid * 16);
        wait_async_zero();
    }
    __syncthreads();

    const int wave = tid >> 5;        // wave32
    const int lane = tid & 31;
    const int g    = lane >> 4;       // lane half-group (K-split in A/B layouts)
    const int m    = lane & 15;       // node row within tile (A side)
    const int n    = lane & 15;       // output column within tile (B/D side)

    const int tile = blockIdx.x * 8 + wave;
    _Float16* Cw = (_Float16*)(smem + SMEM_C_OFF) + wave * (16 * 64);

    // ---- Stage this wave's C tile (16 x 64 f32 -> f16) ----
    const float* Csrc = Cin + (size_t)tile * (16 * 64);
    __builtin_prefetch(Csrc, 0, 0);   // global_prefetch_b8
    for (int i = lane; i < 256; i += 32) {            // 256 float4s
        v4f v = *(const v4f*)(Csrc + i * 4);
        v4h h4 = { (_Float16)v.x, (_Float16)v.y, (_Float16)v.z, (_Float16)v.w };
        *(v4h*)(Cw + i * 4) = h4;
    }
    __syncthreads();

    v8f acc0 = {};   // D tile for k = 0..15
    v8f acc1 = {};   // D tile for k = 16..31

    // ---- Bilinear term: 128 K-chunks of 32 over jl = j*64 + l ----
    // Software-pipelined: next chunk's A-row registers are fetched before the
    // current wmmas so no VALU write overlaps an in-flight wmma source (avoids
    // the 4-slot WMMA->VALU WAR hazard padding).
    _Float16 cj_c = Cw[m * 64 + 0];
    v8h alo_c = *(const v8h*)(Cw + m * 64 + 0 + g * 16);
    v8h ahi_c = *(const v8h*)(Cw + m * 64 + 8 + g * 16);

    #pragma unroll 4
    for (int ch = 0; ch < 128; ++ch) {
        const int kb = ch << 5;         // jl chunk base
        const int j  = kb >> 6;         // constant within chunk
        const int lb = kb & 32;         // l base: 0 or 32

        // B fragments (32x16 f16): elem e -> row K = g*16 + e, col n.
        // B[K, n] = V[k, j, lb + g*16 + e] -> 16 contiguous f16 per lane.
        v16h B0 = *(const v16h*)(Vh + ((size_t)(n)      * 64 + j) * 64 + lb + g * 16);
        v16h B1 = *(const v16h*)(Vh + ((size_t)(n + 16) * 64 + j) * 64 + lb + g * 16);

        // A fragment (16x32 f16): elem e -> l = lb + g*16 + e
        v8h cjv  = { cj_c, cj_c, cj_c, cj_c, cj_c, cj_c, cj_c, cj_c };
        v8h p_lo = alo_c * cjv;         // G[m,jl] = C[m,j]*C[m,l]  (v_pk_mul_f16)
        v8h p_hi = ahi_c * cjv;
        v16h A = __builtin_shufflevector(p_lo, p_hi,
                 0, 1, 2, 3, 4, 5, 6, 7, 8, 9, 10, 11, 12, 13, 14, 15);

        // Prefetch next chunk's A rows into fresh registers (independent work).
        if (ch < 127) {
            const int kb2 = (ch + 1) << 5;
            const int j2  = kb2 >> 6;
            const int lb2 = kb2 & 32;
            cj_c  = Cw[m * 64 + j2];
            alo_c = *(const v8h*)(Cw + m * 64 + lb2 + g * 16);
            ahi_c = *(const v8h*)(Cw + m * 64 + lb2 + 8 + g * 16);
        }

        acc0 = __builtin_amdgcn_wmma_f32_16x16x32_f16(false, A, false, B0,
                                                      (short)0, acc0, false, false);
        acc1 = __builtin_amdgcn_wmma_f32_16x16x32_f16(false, A, false, B1,
                                                      (short)0, acc1, false, false);
    }

    // ---- Linear term c@W: 2 K-chunks of 32 over l, A = C directly ----
    #pragma unroll
    for (int lc = 0; lc < 2; ++lc) {
        const int lb = lc << 5;
        v8h a_lo = *(const v8h*)(Cw + m * 64 + lb + g * 16);
        v8h a_hi = *(const v8h*)(Cw + m * 64 + lb + 8 + g * 16);
        v16h A = __builtin_shufflevector(a_lo, a_hi,
                 0, 1, 2, 3, 4, 5, 6, 7, 8, 9, 10, 11, 12, 13, 14, 15);
        v16h B0 = *(const v16h*)(Wt + (size_t)(n)      * 64 + lb + g * 16);
        v16h B1 = *(const v16h*)(Wt + (size_t)(n + 16) * 64 + lb + g * 16);
        acc0 = __builtin_amdgcn_wmma_f32_16x16x32_f16(false, A, false, B0,
                                                      (short)0, acc0, false, false);
        acc1 = __builtin_amdgcn_wmma_f32_16x16x32_f16(false, A, false, B1,
                                                      (short)0, acc1, false, false);
    }

    // ---- Epilogue: bias + tanh, store. D layout: vgpr r -> row (r + 8g), col n ----
    const float b0 = bias[n];
    const float b1 = bias[16 + n];
    float* out = Hout + (size_t)tile * 16 * 32;
    #pragma unroll
    for (int r = 0; r < 8; ++r) {
        const int mm = r + 8 * g;
        out[mm * 32 + n]      = tanhf(acc0[r] + b0);
        out[mm * 32 + 16 + n] = tanhf(acc1[r] + b1);
    }
}

// ---------------------------------------------------------------------------
// logits = nodes @ Wout_w^T + Wout_b ; log_softmax over 5 classes
// ---------------------------------------------------------------------------
__global__ void rntn_out_kernel(const float* __restrict__ nodes,
                                const float* __restrict__ Ww,
                                const float* __restrict__ Wb,
                                float* __restrict__ lp) {
    int r = blockIdx.x * blockDim.x + threadIdx.x;
    if (r >= NODES_TOTAL) return;
    const float* nr = nodes + (size_t)r * 32;
    float x[OUT_CLS];
    #pragma unroll
    for (int o = 0; o < OUT_CLS; ++o) {
        float s = Wb[o];
        #pragma unroll
        for (int d = 0; d < 32; ++d) s += nr[d] * Ww[o * 32 + d];
        x[o] = s;
    }
    float mx = x[0];
    #pragma unroll
    for (int o = 1; o < OUT_CLS; ++o) mx = fmaxf(mx, x[o]);
    float se = 0.f;
    #pragma unroll
    for (int o = 0; o < OUT_CLS; ++o) se += __expf(x[o] - mx);
    float lse = mx + __logf(se);
    #pragma unroll
    for (int o = 0; o < OUT_CLS; ++o) lp[(size_t)r * OUT_CLS + o] = x[o] - lse;
}

__global__ void rntn_roots_kernel(const float* __restrict__ src,
                                  float* __restrict__ dst) {
    int i = blockIdx.x * blockDim.x + threadIdx.x;   // 512*32 = 16384
    dst[i] = src[i];
}

// ---------------------------------------------------------------------------
extern "C" void kernel_launch(void* const* d_in, const int* in_sizes, int n_in,
                              void* d_out, int out_size, void* d_ws, size_t ws_size,
                              hipStream_t stream) {
    const int*   word_ids = (const int*)  d_in[0];   // (512, 64)
    const float* embed    = (const float*)d_in[1];   // (50000, 32)
    const float* V        = (const float*)d_in[2];   // (32, 64, 64)
    const float* W        = (const float*)d_in[3];   // (64, 32)
    const float* bias     = (const float*)d_in[4];   // (1, 32)
    const float* Wout_w   = (const float*)d_in[5];   // (5, 32)
    const float* Wout_b   = (const float*)d_in[6];   // (5,)
    (void)in_sizes; (void)n_in; (void)out_size; (void)ws_size;

    float* log_probs = (float*)d_out;                                 // 65024*5
    float* roots     = (float*)d_out + (size_t)NODES_TOTAL * OUT_CLS; // 512*32
    float*     nodes = (float*)d_ws;                                  // 65024*32 f32
    _Float16*  Vh_g  = (_Float16*)((char*)d_ws + WS_VH_OFF);
    _Float16*  Wt_g  = (_Float16*)((char*)d_ws + WS_WT_OFF);

    // f16 images of V and W^T (once per call).
    rntn_prep_kernel<<<(32 * 64 * 64) / 256, 256, 0, stream>>>(V, W, Vh_g, Wt_g);

    // Leaves: embedding gather (32768 rows).
    rntn_gather_kernel<<<(LEAF_ROWS * 8) / 256, 256, 0, stream>>>(word_ids, embed, nodes);

    // Six merge levels. Input block viewed as (rows_out, 64) is contiguous
    // because c = concat(h[2i], h[2i+1]) along the row axis.
    int rows_in = LEAF_ROWS;
    int off_in  = 0;
    for (int l = 0; l < 6; ++l) {
        int rows_out = rows_in >> 1;
        int off_out  = off_in + rows_in;
        int tiles    = rows_out / 16;          // 1024,512,256,128,64,32
        rntn_level_kernel<<<tiles / 8, 256, SMEM_TOTAL, stream>>>(
            nodes + (size_t)off_in * 32,
            nodes + (size_t)off_out * 32,
            Vh_g, Wt_g, bias);
        off_in  = off_out;
        rows_in = rows_out;
    }

    // Classifier over all 65024 nodes.
    rntn_out_kernel<<<(NODES_TOTAL + 255) / 256, 256, 0, stream>>>(
        nodes, Wout_w, Wout_b, log_probs);

    // Roots = final level block (last 512 rows).
    rntn_roots_kernel<<<16384 / 256, 256, 0, stream>>>(
        nodes + (size_t)(NODES_TOTAL - 512) * 32, roots);
}
